// KeyedLSTM_30537217475009
// MI455X (gfx1250) — matrix operand order repaired and checked by reading, not compiled, not run on hardware
//
#include <hip/hip_runtime.h>
#include <hip/hip_bf16.h>

// ---------------------------------------------------------------------------
// KeyedLSTM for MI455X (gfx1250): fp16 WMMA recurrence, per-step kernels.
//   B=32 S=256 I=1024 H=2048 KB=4 KL=16 ; K_total = I+H = 3072 ; N = 4H = 8192
// Round 2: conditional-free pipelined k-loops; h staged per-block in LDS in
// fragment layout so only B streams from L2 inside the WMMA loop.
// ---------------------------------------------------------------------------

typedef _Float16 half8  __attribute__((ext_vector_type(8)));
typedef _Float16 half16 __attribute__((ext_vector_type(16)));
typedef float    float8 __attribute__((ext_vector_type(8)));

#define B_   32
#define S_   256
#define I_   1024
#define H_   2048
#define KB_  4
#define KL_  16
#define KTOT 3072              // I_ + H_
#define NG   8192              // 4*H_
#define NT   512               // NG/16  (n tiles)
#define KT   96                // KTOT/32 (k tiles)
#define KTX  32                // x-region k tiles (I_/32)
#define KTH  64                // h-region k tiles (H_/32)

// ---- workspace layout (bytes) ---------------------------------------------
static constexpr size_t SZ_BT   = (size_t)NT * KT * 512 * 2;   // tiled weights fp16
static constexpr size_t SZ_XH   = (size_t)B_ * S_ * I_ * 2;    // x fp16 [s][b][k]
static constexpr size_t SZ_KHX  = (size_t)KL_ * 16 * I_ * 2;   // key x fp16 padded
static constexpr size_t SZ_BIAS = (size_t)4 * H_ * 4;          // fused bias fp32
static constexpr size_t SZ_HB   = (size_t)2 * B_ * H_ * 2;     // h double-buffer
static constexpr size_t SZ_CB   = (size_t)B_ * H_ * 4;         // c fp32
static constexpr size_t SZ_KHB  = (size_t)2 * 16 * H_ * 2;     // key h dbl-buffer
static constexpr size_t SZ_KCB  = (size_t)16 * H_ * 4;         // key c fp32

static constexpr size_t OFF_BT   = 0;
static constexpr size_t OFF_XH   = OFF_BT + SZ_BT;
static constexpr size_t OFF_KHX  = OFF_XH + SZ_XH;
static constexpr size_t OFF_BIAS = OFF_KHX + SZ_KHX;
static constexpr size_t OFF_HB   = OFF_BIAS + SZ_BIAS;
static constexpr size_t OFF_CB   = OFF_HB + SZ_HB;
static constexpr size_t OFF_KHB  = OFF_CB + SZ_CB;
static constexpr size_t OFF_KCB  = OFF_KHB + SZ_KHB;
static constexpr size_t OFF_GM   = OFF_KCB + SZ_KCB;           // gmeans fp32

// ---- fragment helpers ------------------------------------------------------
__device__ __forceinline__ half16 combine16(half8 lo, half8 hi) {
  return __builtin_shufflevector(lo, hi, 0,1,2,3,4,5,6,7,8,9,10,11,12,13,14,15);
}
// B fragment: 16 contiguous halfs (tiled layout [nTile][kTile][lane][16])
__device__ __forceinline__ half16 ldB16(const _Float16* p) {
  return combine16(*(const half8*)p, *(const half8*)(p + 8));
}
// A fragment (16-bit A 16x32): per lane 8 halfs at +0..7 and 8 halfs at +16..23
__device__ __forceinline__ half16 ldA16(const _Float16* p) {
  return combine16(*(const half8*)p, *(const half8*)(p + 16));
}
__device__ __forceinline__ float sig_(float x) { return 1.0f / (1.0f + __expf(-x)); }

__device__ __forceinline__ float8 wmma16(half16 a, half16 b, float8 c) {
  return __builtin_amdgcn_wmma_f32_16x16x32_f16(false, a, false, b, (short)0, c,
                                                false, false);
}

// ---------------------------------------------------------------------------
// prep: x (B,S,I) fp32 -> xh fp16 laid out [s][b][k]
__global__ __launch_bounds__(256) void prep_xh_kernel(const float* __restrict__ x,
                                                      _Float16* __restrict__ xh) {
  size_t i = (size_t)blockIdx.x * 256 + threadIdx.x;
  if (i >= (size_t)B_ * S_ * I_) return;
  int k = (int)(i % I_);
  int r = (int)(i / I_);          // r = s*B_ + b
  int b = r % B_;
  int s = r / B_;
  xh[i] = (_Float16)x[((size_t)b * S_ + s) * I_ + k];
}

// prep: tile concatenated weights [W_ih ; W_hh] (row-major over n, K=3072) into
// WMMA-B fragment layout: btile[((nTile*KT + kTile)*32 + lane)*16 + e]
//   n = nTile*16 + (lane&15) ; k = kTile*32 + (lane>>4)*8 + e + (e>>3)*8
__global__ __launch_bounds__(256) void prep_wtile_kernel(const float* __restrict__ wih,
                                                         const float* __restrict__ whh,
                                                         _Float16* __restrict__ btile) {
  int tid = blockIdx.x * 256 + threadIdx.x;   // NT*KT*32 = 1,572,864
  if (tid >= NT * KT * 32) return;
  int lane = tid & 31;
  int kt   = (tid >> 5) % KT;
  int nt   = tid / (32 * KT);
  int n     = nt * 16 + (lane & 15);
  int kbase = kt * 32 + (lane >> 4) * 8;
  _Float16* dst = btile + (size_t)tid * 16;
  if (kbase < I_) {
    const float* src = wih + (size_t)n * I_ + kbase;
#pragma unroll
    for (int e = 0; e < 16; ++e) dst[e] = (_Float16)src[e + (e >> 3) * 8];
  } else {
    const float* src = whh + (size_t)n * H_ + (kbase - I_);
#pragma unroll
    for (int e = 0; e < 16; ++e) dst[e] = (_Float16)src[e + (e >> 3) * 8];
  }
}

// prep: key_seq -> fp16 padded to 16 rows, fuse biases, zero all state buffers
__global__ __launch_bounds__(256) void prep_misc_kernel(const float* __restrict__ key_seq,
                                                        const float* __restrict__ bih,
                                                        const float* __restrict__ bhh,
                                                        _Float16* __restrict__ khx,
                                                        float* __restrict__ bias,
                                                        _Float16* __restrict__ hbuf,
                                                        float* __restrict__ cbuf,
                                                        _Float16* __restrict__ khbuf,
                                                        float* __restrict__ kcbuf) {
  int i = blockIdx.x * 256 + threadIdx.x;     // grid covers 262,144
  if (i < KL_ * 16 * I_) {
    int k   = i % I_;
    int rr  = i / I_;                          // rr = t*16 + row
    int row = rr % 16;
    int t   = rr / 16;
    float v = (row < KB_) ? key_seq[((size_t)row * KL_ + t) * I_ + k] : 0.0f;
    khx[i] = (_Float16)v;
  }
  if (i < 2 * B_ * H_)  hbuf[i]  = (_Float16)0.0f;
  if (i < B_ * H_)      cbuf[i]  = 0.0f;
  if (i < 2 * 16 * H_)  khbuf[i] = (_Float16)0.0f;
  if (i < 16 * H_)      kcbuf[i] = 0.0f;
  if (i < 4 * H_)       bias[i]  = bih[i] + bhh[i];
}

// ---------------------------------------------------------------------------
// Key recurrence step t: gates = [kx_t, kh] @ W^T + bias over M=16 (KB=4 padded)
// 128 WGs x 4 waves; wave j owns gate j for 16 state columns; gmeans written.
__global__ __launch_bounds__(128) void key_step_kernel(const _Float16* __restrict__ khx,
                                                       const _Float16* __restrict__ btile,
                                                       const float* __restrict__ bias,
                                                       _Float16* __restrict__ khbuf,
                                                       float* __restrict__ kcbuf,
                                                       float* __restrict__ gmeans,
                                                       int t) {
  __shared__ float smem[4 * 256 + 256];
  const int tid  = threadIdx.x;
  const int lane = tid & 31;
  const int gate = tid >> 5;                 // wave id == gate
  const int wg   = blockIdx.x;               // 0..127 state-column block
  const int nTile = gate * 128 + wg;
  const int khalf = lane >> 4;
  const int row0  = lane & 15;

  const _Float16* bptr  = btile + ((size_t)nTile * KT) * 512 + lane * 16;
  const _Float16* xp    = khx + ((size_t)t * 16 + row0) * I_ + khalf * 8;
  const _Float16* hread = khbuf + (size_t)(t & 1) * 16 * H_;
  _Float16*       hwrt  = khbuf + (size_t)((t + 1) & 1) * 16 * H_;
  const _Float16* hp    = hread + (size_t)row0 * H_ + khalf * 8;

  float8 acc = {};
  // ---- x region (kt 0..31), conditional-free, pipelined ----
  half16 bf = ldB16(bptr);
  half16 af = ldA16(xp);
  for (int kt = 0; kt < KTX; ++kt) {
    half16 bf_c = bf, af_c = af;
    bptr += 512;
    if (kt + 1 < KTX) { bf = ldB16(bptr); af = ldA16(xp + (kt + 1) * 32); }
    acc = wmma16(af_c, bf_c, acc);
  }
  // ---- h region (kt 32..95) ----
  bf = ldB16(bptr);
  af = ldA16(hp);
  for (int kt = 0; kt < KTH; ++kt) {
    half16 bf_c = bf, af_c = af;
    bptr += 512;
    if (kt + 1 < KTH) { bf = ldB16(bptr); af = ldA16(hp + (kt + 1) * 32); }
    acc = wmma16(af_c, bf_c, acc);
  }
#pragma unroll
  for (int r = 0; r < 8; ++r) smem[gate * 256 + r * 32 + lane] = acc[r];
  __syncthreads();

  for (int e = tid; e < 256; e += 128) {
    int m  = e >> 4, nn = e & 15;
    int idx = (m & 7) * 32 + (m >> 3) * 16 + nn;   // C-layout index
    int col = wg * 16 + nn;
    float vi = smem[0 * 256 + idx] + bias[0 * H_ + col];
    float vf = smem[1 * 256 + idx] + bias[1 * H_ + col];
    float vg = smem[2 * 256 + idx] + bias[2 * H_ + col];
    float vo = smem[3 * 256 + idx] + bias[3 * H_ + col];
    float ii = sig_(vi), ff = sig_(vf), gg = tanhf(vg), oo = sig_(vo);
    size_t sidx = (size_t)m * H_ + col;
    float c = ff * kcbuf[sidx] + ii * gg;
    float h = oo * tanhf(c);
    hwrt[sidx]  = (_Float16)h;
    kcbuf[sidx] = c;
    smem[1024 + m * 16 + nn] = ff;                 // stash forget activation
  }
  __syncthreads();
  if (tid < 16) {   // mean over the KB_=4 real key-batch rows
    float gm = 0.25f * (smem[1024 + 0 * 16 + tid] + smem[1024 + 1 * 16 + tid] +
                        smem[1024 + 2 * 16 + tid] + smem[1024 + 3 * 16 + tid]);
    gmeans[(size_t)t * H_ + wg * 16 + tid] = gm;
  }
}

// ---------------------------------------------------------------------------
// Main recurrence step t. 128 WGs x 8 waves. Wave j: gate=j&3, kpart=j>>2
// (split-K halves of KT=96), each computing a 32x16 fp32 tile via 2 WMMA/k-tile.
// h (32x2048 fp16) is staged once per block into LDS in fragment layout:
//   Ah[mt][kt][lane] = 16 halfs = A-fragment lane data for M-tile mt, h-kTile kt.
// Inside the WMMA loop only B streams from (L2-resident) global memory.
// h double-buffered across steps (cross-block WAR); c per-column-owned.
__global__ __launch_bounds__(256) void lstm_step_kernel(const _Float16* __restrict__ xh,
                                                        const _Float16* __restrict__ btile,
                                                        const float* __restrict__ bias,
                                                        const float* __restrict__ gmeans,
                                                        _Float16* __restrict__ hbuf,
                                                        float* __restrict__ cbuf,
                                                        float* __restrict__ out,
                                                        int t) {
  __shared__ half16 Ah[2][KTH][32];   // 128 KB: h in A-fragment layout
  __shared__ float  red[8 * 512];     // 16 KB : split-K reduction
  const int tid   = threadIdx.x;
  const int lane  = tid & 31;
  const int wave  = tid >> 5;
  const int gate  = wave & 3;
  const int kpart = wave >> 2;
  const int wg    = blockIdx.x;               // 0..127 state-column block
  const int nTile = gate * 128 + wg;
  const int khalf = lane >> 4;
  const int row0  = lane & 15;

  const _Float16* hread = hbuf + (size_t)(t & 1) * B_ * H_;
  _Float16*       hwrt  = hbuf + (size_t)((t + 1) & 1) * B_ * H_;

  // ---- stage h into LDS in fragment layout (2*64*32 entries of 32B) ----
  for (int ent = tid; ent < 2 * KTH * 32; ent += 256) {
    int sl  = ent & 31;
    int skt = (ent >> 5) & (KTH - 1);
    int smt = ent >> 11;
    int srow = smt * 16 + (sl & 15);
    const _Float16* src = hread + (size_t)srow * H_ + skt * 32 + (sl >> 4) * 8;
    Ah[smt][skt][sl] = combine16(*(const half8*)src, *(const half8*)(src + 16));
  }
  __syncthreads();

  const int ktile0 = kpart * (KT / 2);        // 0 or 48
  const _Float16* bptr = btile + ((size_t)nTile * KT + ktile0) * 512 + lane * 16;

  float8 acc0 = {}, acc1 = {};
  if (kpart == 0) {
    // ---- x region: kt 0..31, A from global (pipelined, conditional-free) ----
    const _Float16* xp0 = xh + ((size_t)t * B_ + row0) * I_ + khalf * 8;
    const _Float16* xp1 = xh + ((size_t)t * B_ + 16 + row0) * I_ + khalf * 8;
    half16 bf = ldB16(bptr);
    half16 a0 = ldA16(xp0);
    half16 a1 = ldA16(xp1);
    for (int kt = 0; kt < KTX; ++kt) {
      half16 bf_c = bf, a0_c = a0, a1_c = a1;
      bptr += 512;
      if (kt + 1 < KTX) {
        bf = ldB16(bptr);
        a0 = ldA16(xp0 + (kt + 1) * 32);
        a1 = ldA16(xp1 + (kt + 1) * 32);
      }
      acc0 = wmma16(a0_c, bf_c, acc0);
      acc1 = wmma16(a1_c, bf_c, acc1);
    }
    // ---- h region: kt 32..47 -> LDS tiles 0..15 ----
    bf = ldB16(bptr);
    for (int kt = 0; kt < 16; ++kt) {
      half16 bf_c = bf;
      bptr += 512;
      if (kt + 1 < 16) bf = ldB16(bptr);
      half16 a0h = Ah[0][kt][lane];
      half16 a1h = Ah[1][kt][lane];
      acc0 = wmma16(a0h, bf_c, acc0);
      acc1 = wmma16(a1h, bf_c, acc1);
    }
  } else {
    // ---- h region: kt 48..95 -> LDS tiles 16..63 ----
    half16 bf = ldB16(bptr);
    for (int kt = 16; kt < KTH; ++kt) {
      half16 bf_c = bf;
      bptr += 512;
      if (kt + 1 < KTH) bf = ldB16(bptr);
      half16 a0h = Ah[0][kt][lane];
      half16 a1h = Ah[1][kt][lane];
      acc0 = wmma16(a0h, bf_c, acc0);
      acc1 = wmma16(a1h, bf_c, acc1);
    }
  }

  float* my = red + wave * 512;
#pragma unroll
  for (int r = 0; r < 8; ++r) {
    my[r * 32 + lane]       = acc0[r];
    my[256 + r * 32 + lane] = acc1[r];
  }
  __syncthreads();

  for (int e = tid; e < 512; e += 256) {
    int m  = e >> 4, nn = e & 15;
    int mt = m >> 4, mm = m & 15;
    int idx = mt * 256 + (mm & 7) * 32 + (mm >> 3) * 16 + nn;
    int col = wg * 16 + nn;
    float vi = red[0 * 512 + idx] + red[4 * 512 + idx] + bias[0 * H_ + col];
    float vf = red[1 * 512 + idx] + red[5 * 512 + idx] + bias[1 * H_ + col];
    float vg = red[2 * 512 + idx] + red[6 * 512 + idx] + bias[2 * H_ + col];
    float vo = red[3 * 512 + idx] + red[7 * 512 + idx] + bias[3 * H_ + col];
    float ii = sig_(vi), ff = sig_(vf), gg = tanhf(vg), oo = sig_(vo);
    size_t sidx = (size_t)m * H_ + col;
    float c = ff * cbuf[sidx] + ii * gg;
    float h = oo * tanhf(c);
    out[(size_t)t * B_ * H_ + sidx] = h;               // pre-multiplier output
    float mlt = (t < KL_) ? gmeans[(size_t)t * H_ + col] : 1.0f;
    hwrt[sidx] = (_Float16)(h * mlt);
    cbuf[sidx] = c * mlt;
  }
}

// ---------------------------------------------------------------------------
extern "C" void kernel_launch(void* const* d_in, const int* in_sizes, int n_in,
                              void* d_out, int out_size, void* d_ws, size_t ws_size,
                              hipStream_t stream) {
  (void)in_sizes; (void)n_in; (void)out_size; (void)ws_size;
  const float* x       = (const float*)d_in[0];
  const float* key_seq = (const float*)d_in[1];
  const float* wih     = (const float*)d_in[2];
  const float* whh     = (const float*)d_in[3];
  const float* bih     = (const float*)d_in[4];
  const float* bhh     = (const float*)d_in[5];
  float* out = (float*)d_out;

  char* ws = (char*)d_ws;
  _Float16* btile = (_Float16*)(ws + OFF_BT);
  _Float16* xh    = (_Float16*)(ws + OFF_XH);
  _Float16* khx   = (_Float16*)(ws + OFF_KHX);
  float*    bias  = (float*)(ws + OFF_BIAS);
  _Float16* hbuf  = (_Float16*)(ws + OFF_HB);
  float*    cbuf  = (float*)(ws + OFF_CB);
  _Float16* khbuf = (_Float16*)(ws + OFF_KHB);
  float*    kcbuf = (float*)(ws + OFF_KCB);
  float*    gmean = (float*)(ws + OFF_GM);

  prep_wtile_kernel<<<(NT * KT * 32 + 255) / 256, 256, 0, stream>>>(wih, whh, btile);
  prep_xh_kernel<<<(B_ * S_ * I_ + 255) / 256, 256, 0, stream>>>(x, xh);
  prep_misc_kernel<<<(KL_ * 16 * I_ + 255) / 256, 256, 0, stream>>>(
      key_seq, bih, bhh, khx, bias, hbuf, cbuf, khbuf, kcbuf);

  for (int t = 0; t < KL_; ++t)
    key_step_kernel<<<128, 128, 0, stream>>>(khx, btile, bias, khbuf, kcbuf, gmean, t);

  for (int t = 0; t < S_; ++t)
    lstm_step_kernel<<<128, 256, 0, stream>>>(xh, btile, bias, gmean, hbuf, cbuf, out, t);
}
